// GNN_node_Virtualnode_57062935495534
// MI455X (gfx1250) — matrix operand order repaired and checked by reading, not compiled, not run on hardware
//
#include <hip/hip_runtime.h>
#include <hip/hip_bf16.h>

// ---------------------------------------------------------------------------
// GIN + virtual-node GNN on MI455X (gfx1250, wave32).
// Edge gather/scatter is the memory-bound part (≈5 GB traffic @ 23.3 TB/s);
// the dense MLP GEMMs run on v_wmma_f32_16x16x32_bf16. Each wave owns a
// 16x64 output strip (4 column tiles), so every A fragment feeds 4 WMMAs.
// ---------------------------------------------------------------------------

#define Nn 100000
#define Ee 1600000
#define Gg 128
#define Dd 128
#define Ll 3
#define EAa 7

typedef __attribute__((ext_vector_type(16))) __bf16       v16bf;
typedef __attribute__((ext_vector_type(8)))  float        f32x8;
typedef __attribute__((ext_vector_type(4)))  unsigned int u32x4;

union Frag { v16bf v; u32x4 q[2]; };

__device__ __forceinline__ unsigned short f2bf(float f) {
    unsigned int u = __float_as_uint(f);
    unsigned int r = u + 0x7FFFu + ((u >> 16) & 1u);   // round-to-nearest-even
    return (unsigned short)(r >> 16);
}

// ---------------- small elementwise kernels --------------------------------

__global__ void k_init_h(const int* __restrict__ x, const float* __restrict__ node_emb,
                         float* __restrict__ h) {
    long long i = (long long)blockIdx.x * blockDim.x + threadIdx.x;
    if (i >= (long long)Nn * Dd) return;
    int node = (int)(i >> 7), d = (int)(i & 127);
    h[i] = node_emb[(long long)x[node] * Dd + d];
}

__global__ void k_init_vn(const float* __restrict__ vn_emb, float* __restrict__ vn) {
    int i = blockIdx.x * blockDim.x + threadIdx.x;
    if (i >= Gg * Dd) return;
    vn[i] = vn_emb[i & 127];
}

// h_in = h + vn[batch];  agg = (1+eps[l]) * h_in   (agg later gets edge msgs)
__global__ void k_node_prep(const float* __restrict__ h, const float* __restrict__ vn,
                            const int* __restrict__ batch, const float* __restrict__ eps,
                            int l, float* __restrict__ h_in, float* __restrict__ agg) {
    long long i = (long long)blockIdx.x * blockDim.x + threadIdx.x;
    if (i >= (long long)Nn * Dd) return;
    int node = (int)(i >> 7), d = (int)(i & 127);
    float v = h[i] + vn[(long long)batch[node] * Dd + d];
    h_in[i] = v;
    agg[i] = (1.0f + eps[l]) * v;
}

__global__ void k_f32_to_bf16(const float* __restrict__ src, unsigned short* __restrict__ dst,
                              long long n) {
    long long i = (long long)blockIdx.x * blockDim.x + threadIdx.x;
    if (i >= n) return;
    dst[i] = f2bf(src[i]);
}

__global__ void k_copy(const float* __restrict__ src, float* __restrict__ dst, int n) {
    int i = blockIdx.x * blockDim.x + threadIdx.x;
    if (i < n) dst[i] = src[i];
}

// vt[batch[i]][d] += h_in[i][d]   (global_add_pool)
__global__ void k_vn_accum(const float* __restrict__ h_in, const int* __restrict__ batch,
                           float* __restrict__ vt) {
    long long i = (long long)blockIdx.x * blockDim.x + threadIdx.x;
    if (i >= (long long)Nn * Dd) return;
    int node = (int)(i >> 7), d = (int)(i & 127);
    __hip_atomic_fetch_add(&vt[(long long)batch[node] * Dd + d], h_in[i],
                           __ATOMIC_RELAXED, __HIP_MEMORY_SCOPE_AGENT);
}

// ---------------- edge kernel: e = attr@W + b; msg = relu(h_in[src]+e);
// agg[dst] += msg  -----------------------------------------------------------

__global__ void k_edges(const float* __restrict__ edge_attr, const int* __restrict__ src,
                        const int* __restrict__ dst, const float* __restrict__ Wl,
                        const float* __restrict__ bl, const float* __restrict__ h_in,
                        float* __restrict__ agg) {
    __shared__ float sW[EAa * Dd];
    __shared__ float sB[Dd];
    for (int i = threadIdx.x; i < EAa * Dd; i += blockDim.x) sW[i] = Wl[i];
    for (int i = threadIdx.x; i < Dd; i += blockDim.x) sB[i] = bl[i];
    __syncthreads();
    int d   = threadIdx.x & 127;
    int sub = threadIdx.x >> 7;            // which edge within the block slice
    int per = blockDim.x >> 7;             // edges per block per iteration
    for (long long e = (long long)blockIdx.x * per + sub; e < Ee;
         e += (long long)gridDim.x * per) {
        float v = sB[d];
        const float* a = edge_attr + e * EAa;
#pragma unroll
        for (int k = 0; k < EAa; ++k) v = fmaf(a[k], sW[k * Dd + d], v);
        int s = src[e], t = dst[e];
        float msg = fmaxf(h_in[(long long)s * Dd + d] + v, 0.0f);
        __hip_atomic_fetch_add(&agg[(long long)t * Dd + d], msg,
                               __ATOMIC_RELAXED, __HIP_MEMORY_SCOPE_AGENT);
    }
}

// ---------------- weight pre-pack into WMMA B-fragment order ---------------
// packed idx = ((nt*Kc + kc)*32 + lane)*16 + e
// element (lane, e) of chunk kc holds W[kc*32 + ((e&8)?16:0) + ((lane&16)?8:0) + (e&7)][nt*16 + (lane&15)]

__global__ void k_prepack(const float* __restrict__ W, unsigned short* __restrict__ Bp,
                          int K, int Nc) {
    int idx = blockIdx.x * blockDim.x + threadIdx.x;
    if (idx >= K * Nc) return;
    int e    = idx & 15;
    int lane = (idx >> 4) & 31;
    int rem  = idx >> 9;                   // nt*Kc + kc
    int Kc   = K >> 5;
    int kc   = rem % Kc;
    int nt   = rem / Kc;
    int kk   = kc * 32 + ((e & 8) ? 16 : 0) + ((lane & 16) ? 8 : 0) + (e & 7);
    int col  = nt * 16 + (lane & 15);
    Bp[idx] = f2bf(W[(long long)kk * Nc + col]);
}

// ---------------- WMMA GEMM + bias + BN + optional ReLU --------------------
// C[M,Nc] = A[M,K](bf16,row-major) x Bp(packed bf16).
// One wave computes a 16x64 strip: 4 column tiles, 4x v8f accumulators; each
// A fragment is shared by 4 back-to-back v_wmma ops (A traffic cut 4x).

__global__ void k_wmma_gemm(const unsigned short* __restrict__ A,
                            const unsigned short* __restrict__ Bp,
                            const float* __restrict__ bias,
                            const float* __restrict__ bn_g, const float* __restrict__ bn_b,
                            const float* __restrict__ bn_m, const float* __restrict__ bn_v,
                            float* __restrict__ outF, unsigned short* __restrict__ outB,
                            int M, int K, int Nc, int relu) {
    int wid  = blockIdx.x * (blockDim.x >> 5) + (threadIdx.x >> 5);
    int lane = threadIdx.x & 31;
    int ngroups = Nc >> 6;                     // 64 columns per wave
    int total   = (M >> 4) * ngroups;
    if (wid >= total) return;                  // wave-uniform: EXEC stays all-1s
    int ng = wid % ngroups, tm = wid / ngroups;
    int m0 = tm << 4;
    int Kc = K >> 5;

    int row    = m0 + (lane & 15);
    int kbaseA = (lane & 16) ? 8 : 0;          // 16-bit A layout (ISA 7.12.2)
    const unsigned short* Arow = A + (long long)row * K + kbaseA;
    const long long tstride = (long long)Kc * 512;          // elems per n-tile
    const unsigned short* Bt = Bp + (long long)(ng * 4) * tstride + (long long)lane * 16;

    f32x8 acc0 = {}, acc1 = {}, acc2 = {}, acc3 = {};
    for (int kc = 0; kc < Kc; ++kc) {
        Frag a;
        a.q[0] = *(const u32x4*)(Arow + kc * 32);        // K = kbase+0..7
        a.q[1] = *(const u32x4*)(Arow + kc * 32 + 16);   // K = kbase+16..23
        const unsigned short* Bc = Bt + kc * 512;
        Frag b0, b1, b2, b3;
        b0.q[0] = *(const u32x4*)(Bc);
        b0.q[1] = *(const u32x4*)(Bc + 8);
        b1.q[0] = *(const u32x4*)(Bc + tstride);
        b1.q[1] = *(const u32x4*)(Bc + tstride + 8);
        b2.q[0] = *(const u32x4*)(Bc + 2 * tstride);
        b2.q[1] = *(const u32x4*)(Bc + 2 * tstride + 8);
        b3.q[0] = *(const u32x4*)(Bc + 3 * tstride);
        b3.q[1] = *(const u32x4*)(Bc + 3 * tstride + 8);
        if (kc + 1 < Kc) __builtin_prefetch(Arow + (kc + 1) * 32, 0, 1);
        acc0 = __builtin_amdgcn_wmma_f32_16x16x32_bf16(false, a.v, false, b0.v,
                                                       (short)0, acc0, false, false);
        acc1 = __builtin_amdgcn_wmma_f32_16x16x32_bf16(false, a.v, false, b1.v,
                                                       (short)0, acc1, false, false);
        acc2 = __builtin_amdgcn_wmma_f32_16x16x32_bf16(false, a.v, false, b2.v,
                                                       (short)0, acc2, false, false);
        acc3 = __builtin_amdgcn_wmma_f32_16x16x32_bf16(false, a.v, false, b3.v,
                                                       (short)0, acc3, false, false);
    }

    // epilogue: C layout (ISA 7.12.2) — lane 0-15: rows m0+0..7, lane 16-31: rows m0+8..15
    int rbase = (lane & 16) ? 8 : 0;
    f32x8 accs[4] = {acc0, acc1, acc2, acc3};
#pragma unroll
    for (int t = 0; t < 4; ++t) {
        int col = (ng << 6) + (t << 4) + (lane & 15);
        float sc = bn_g[col] * __frsqrt_rn(bn_v[col] + 1e-5f);
        float sh = bn_b[col] - bn_m[col] * sc;
        float bs = bias[col];
#pragma unroll
        for (int r = 0; r < 8; ++r) {
            float val = (accs[t][r] + bs) * sc + sh;
            if (relu) val = fmaxf(val, 0.0f);
            long long off = (long long)(m0 + rbase + r) * Nc + col;
            if (outF) outF[off] = val;
            if (outB) outB[off] = f2bf(val);
        }
    }
}

// ---------------------------------------------------------------------------

extern "C" void kernel_launch(void* const* d_in, const int* in_sizes, int n_in,
                              void* d_out, int out_size, void* d_ws, size_t ws_size,
                              hipStream_t stream) {
    (void)in_sizes; (void)n_in; (void)out_size; (void)ws_size;
    const int*   x         = (const int*)d_in[0];
    const int*   ei        = (const int*)d_in[1];
    const float* edge_attr = (const float*)d_in[2];
    const int*   batch     = (const int*)d_in[3];
    const float* node_emb  = (const float*)d_in[4];
    const float* vn_emb    = (const float*)d_in[5];
    const float* edge_W    = (const float*)d_in[6];
    const float* edge_b    = (const float*)d_in[7];
    const float* eps       = (const float*)d_in[8];
    const float* mlp_W1    = (const float*)d_in[9];
    const float* mlp_b1    = (const float*)d_in[10];
    const float* mbn_g     = (const float*)d_in[11];
    const float* mbn_b     = (const float*)d_in[12];
    const float* mbn_m     = (const float*)d_in[13];
    const float* mbn_v     = (const float*)d_in[14];
    const float* mlp_W2    = (const float*)d_in[15];
    const float* mlp_b2    = (const float*)d_in[16];
    const float* bn_g      = (const float*)d_in[17];
    const float* bn_b      = (const float*)d_in[18];
    const float* bn_m      = (const float*)d_in[19];
    const float* bn_v      = (const float*)d_in[20];
    const float* vn_W1     = (const float*)d_in[21];
    const float* vn_b1     = (const float*)d_in[22];
    const float* vbn1_g    = (const float*)d_in[23];
    const float* vbn1_b    = (const float*)d_in[24];
    const float* vbn1_m    = (const float*)d_in[25];
    const float* vbn1_v    = (const float*)d_in[26];
    const float* vn_W2     = (const float*)d_in[27];
    const float* vn_b2     = (const float*)d_in[28];
    const float* vbn2_g    = (const float*)d_in[29];
    const float* vbn2_b    = (const float*)d_in[30];
    const float* vbn2_m    = (const float*)d_in[31];
    const float* vbn2_v    = (const float*)d_in[32];
    float* out = (float*)d_out;

    // --- workspace carve-out (aligned) ---
    char* ws = (char*)d_ws;
    auto alloc = [&](size_t bytes) -> char* {
        char* p = ws;
        ws += (bytes + 255) & ~(size_t)255;
        return p;
    };
    float*          h     = (float*)alloc((size_t)Nn * Dd * 4);
    float*          h_in  = (float*)alloc((size_t)Nn * Dd * 4);
    float*          agg   = (float*)alloc((size_t)Nn * Dd * 4);
    unsigned short* Abf   = (unsigned short*)alloc((size_t)Nn * Dd * 2);
    unsigned short* z_bf  = (unsigned short*)alloc((size_t)Nn * 2 * Dd * 2);
    float*          vn    = (float*)alloc((size_t)Gg * Dd * 4);
    float*          vt    = (float*)alloc((size_t)Gg * Dd * 4);
    unsigned short* vt_bf = (unsigned short*)alloc((size_t)Gg * Dd * 2);
    unsigned short* t_bf  = (unsigned short*)alloc((size_t)Gg * 2 * Dd * 2);
    unsigned short* W1p   = (unsigned short*)alloc((size_t)Ll * Dd * 2 * Dd * 2);
    unsigned short* W2p   = (unsigned short*)alloc((size_t)Ll * 2 * Dd * Dd * 2);
    unsigned short* vW1p  = (unsigned short*)alloc((size_t)(Ll - 1) * Dd * 2 * Dd * 2);
    unsigned short* vW2p  = (unsigned short*)alloc((size_t)(Ll - 1) * 2 * Dd * Dd * 2);

    const int*  src = ei;
    const int*  dst = ei + Ee;
    const long long ND = (long long)Nn * Dd;
    const int   blkND = (int)((ND + 255) / 256);
    const int   WSZ = Dd * 2 * Dd;   // 32768 elements per weight matrix

    // --- pre-pack all weight matrices into WMMA B-fragment order ---
    for (int l = 0; l < Ll; ++l) {
        k_prepack<<<WSZ / 256, 256, 0, stream>>>(mlp_W1 + (size_t)l * WSZ, W1p + (size_t)l * WSZ, Dd, 2 * Dd);
        k_prepack<<<WSZ / 256, 256, 0, stream>>>(mlp_W2 + (size_t)l * WSZ, W2p + (size_t)l * WSZ, 2 * Dd, Dd);
    }
    for (int l = 0; l < Ll - 1; ++l) {
        k_prepack<<<WSZ / 256, 256, 0, stream>>>(vn_W1 + (size_t)l * WSZ, vW1p + (size_t)l * WSZ, Dd, 2 * Dd);
        k_prepack<<<WSZ / 256, 256, 0, stream>>>(vn_W2 + (size_t)l * WSZ, vW2p + (size_t)l * WSZ, 2 * Dd, Dd);
    }

    // --- init node features and virtual node ---
    k_init_h<<<blkND, 256, 0, stream>>>(x, node_emb, h);
    k_init_vn<<<(Gg * Dd + 255) / 256, 256, 0, stream>>>(vn_emb, vn);

    for (int l = 0; l < Ll; ++l) {
        // h_in = h + vn[batch]; agg = (1+eps)*h_in
        k_node_prep<<<blkND, 256, 0, stream>>>(h, vn, batch, eps, l, h_in, agg);
        // edge messages + scatter-add
        k_edges<<<8192, 256, 0, stream>>>(edge_attr, src, dst,
                                          edge_W + (size_t)l * EAa * Dd,
                                          edge_b + (size_t)l * Dd, h_in, agg);
        // GIN MLP: z = relu(bn(agg @ W1 + b1)); h2 = bn(z @ W2 + b2)[, relu]
        k_f32_to_bf16<<<blkND, 256, 0, stream>>>(agg, Abf, ND);
        {
            int waves = (Nn / 16) * (2 * Dd / 64);      // 25000 strips of 16x64
            k_wmma_gemm<<<(waves + 3) / 4, 128, 0, stream>>>(
                Abf, W1p + (size_t)l * WSZ, mlp_b1 + (size_t)l * 2 * Dd,
                mbn_g + (size_t)l * 2 * Dd, mbn_b + (size_t)l * 2 * Dd,
                mbn_m + (size_t)l * 2 * Dd, mbn_v + (size_t)l * 2 * Dd,
                nullptr, z_bf, Nn, Dd, 2 * Dd, 1);
        }
        {
            int waves = (Nn / 16) * (Dd / 64);          // 12500
            float* dest = (l == Ll - 1) ? out : h;
            k_wmma_gemm<<<(waves + 3) / 4, 128, 0, stream>>>(
                z_bf, W2p + (size_t)l * WSZ, mlp_b2 + (size_t)l * Dd,
                bn_g + (size_t)l * Dd, bn_b + (size_t)l * Dd,
                bn_m + (size_t)l * Dd, bn_v + (size_t)l * Dd,
                dest, nullptr, Nn, 2 * Dd, Dd, (l < Ll - 1) ? 1 : 0);
        }
        // virtual-node update
        if (l < Ll - 1) {
            k_copy<<<(Gg * Dd + 255) / 256, 256, 0, stream>>>(vn, vt, Gg * Dd);
            k_vn_accum<<<blkND, 256, 0, stream>>>(h_in, batch, vt);
            k_f32_to_bf16<<<(Gg * Dd + 255) / 256, 256, 0, stream>>>(vt, vt_bf, (long long)Gg * Dd);
            {
                int waves = (Gg / 16) * (2 * Dd / 64);  // 32
                k_wmma_gemm<<<(waves + 3) / 4, 128, 0, stream>>>(
                    vt_bf, vW1p + (size_t)l * WSZ, vn_b1 + (size_t)l * 2 * Dd,
                    vbn1_g + (size_t)l * 2 * Dd, vbn1_b + (size_t)l * 2 * Dd,
                    vbn1_m + (size_t)l * 2 * Dd, vbn1_v + (size_t)l * 2 * Dd,
                    nullptr, t_bf, Gg, Dd, 2 * Dd, 1);
            }
            {
                int waves = (Gg / 16) * (Dd / 64);      // 16
                k_wmma_gemm<<<(waves + 3) / 4, 128, 0, stream>>>(
                    t_bf, vW2p + (size_t)l * WSZ, vn_b2 + (size_t)l * Dd,
                    vbn2_g + (size_t)l * Dd, vbn2_b + (size_t)l * Dd,
                    vbn2_m + (size_t)l * Dd, vbn2_v + (size_t)l * Dd,
                    vn, nullptr, Gg, 2 * Dd, Dd, 1);
            }
        }
    }
}
